// RegularSimilar_47150150975778
// MI455X (gfx1250) — matrix operand order repaired and checked by reading, not compiled, not run on hardware
//
#include <hip/hip_runtime.h>
#include <hip/hip_bf16.h>
#include <stdint.h>

// ---------------------------------------------------------------------------
// Problem constants (reference: B=2048, N=50000, D=128)
// ---------------------------------------------------------------------------
#define BATCH   2048
#define NITEMS  50000
#define DFEAT   128
#define KIN     256              // 2*D
#define NTILES  (NITEMS / 16)    // 3125 tiles of 16 items
#define NSPLIT  125              // N-splits; 3125 = 125 * 25
#define TILES_PER_SPLIT (NTILES / NSPLIT) // 25 (odd: 12 pipelined pairs + 1 tail)
#define LEAKY   0.01f
#define BN_EPS_ 1e-5f
#define COS_EPS_ 1e-6f
#define NLN2    0.69314718056f   // ln(2)

// ---------------------------------------------------------------------------
// Workspace layout (bytes). Total ~14.4 MB.
// ---------------------------------------------------------------------------
#define WS_H_F32      0u
#define WS_H_BF16     (BATCH * DFEAT * 4u)
#define WS_ITEMS_BF16 (WS_H_BF16 + BATCH * DFEAT * 2u)
#define WS_BEST       (WS_ITEMS_BF16 + NITEMS * DFEAT * 2u)
#define WS_SIM        (WS_BEST + BATCH * 8u)

typedef __attribute__((ext_vector_type(16))) __bf16 v16bf;
typedef __attribute__((ext_vector_type(8)))  float  v8f;

__device__ __forceinline__ unsigned short f2bf(float f) {
    unsigned int u = __float_as_uint(f);
    u += 0x7fffu + ((u >> 16) & 1u);     // round-to-nearest-even
    return (unsigned short)(u >> 16);
}

__device__ __forceinline__ unsigned avalanche(unsigned h) {
    h ^= h >> 16; h *= 0x7FEB352Du;
    h ^= h >> 15; h *= 0x846CA68Bu;
    h ^= h >> 16;
    return h;
}

// ---------------------------------------------------------------------------
// Kernel 1: h_f32 = item_feature(2048x256) @ W(256x128) + b  (f32 for BN stats)
// ---------------------------------------------------------------------------
__global__ void k_mlp(const float* __restrict__ x, const float* __restrict__ W,
                      const float* __restrict__ bias, float* __restrict__ h) {
    int g   = blockIdx.x * blockDim.x + threadIdx.x;
    int row = g >> 7;
    int col = g & 127;
    const float* xr = x + row * KIN;
    float acc = bias[col];
#pragma unroll 8
    for (int k = 0; k < KIN; ++k)
        acc = fmaf(xr[k], W[k * DFEAT + col], acc);
    h[g] = acc;
}

// ---------------------------------------------------------------------------
// Kernel 2: per-column batchnorm + leaky-relu + bf16 convert
// ---------------------------------------------------------------------------
__global__ void k_bn(const float* __restrict__ h, const float* __restrict__ gamma,
                     const float* __restrict__ beta, unsigned short* __restrict__ hb) {
    int col = blockIdx.x;
    int t   = threadIdx.x;                  // 256 threads
    __shared__ float s1[256], s2[256];
    float a = 0.f, b = 0.f;
    for (int r = t; r < BATCH; r += 256) {
        float v = h[r * DFEAT + col];
        a += v; b += v * v;
    }
    s1[t] = a; s2[t] = b;
    __syncthreads();
    for (int o = 128; o > 0; o >>= 1) {
        if (t < o) { s1[t] += s1[t + o]; s2[t] += s2[t + o]; }
        __syncthreads();
    }
    float mean  = s1[0] * (1.f / BATCH);
    float var   = s2[0] * (1.f / BATCH) - mean * mean;
    float scale = rsqrtf(var + BN_EPS_) * gamma[col];
    float shift = beta[col] - mean * scale;
    for (int r = t; r < BATCH; r += 256) {
        float v = h[r * DFEAT + col] * scale + shift;
        v = (v >= 0.f) ? v : LEAKY * v;
        hb[r * DFEAT + col] = f2bf(v);
    }
}

// ---------------------------------------------------------------------------
// Kernel 3: bf16 copy of all_items
// ---------------------------------------------------------------------------
__global__ void k_cvt_items(const float* __restrict__ items, unsigned short* __restrict__ out) {
    int g = blockIdx.x * blockDim.x + threadIdx.x;
    out[g] = f2bf(items[g]);
}

// ---------------------------------------------------------------------------
// Kernel 4: reset packed argmax accumulators (re-zeroed every call)
// ---------------------------------------------------------------------------
__global__ void k_init(unsigned long long* __restrict__ best) {
    int g = blockIdx.x * blockDim.x + threadIdx.x;
    if (g < BATCH) best[g] = 0ull;
}

// ---------------------------------------------------------------------------
// Kernel 5: fused scores = h @ all_items.T + gumbel + argmax (bf16 WMMA)
// ---------------------------------------------------------------------------
union Frag { v16bf v; uint4 u[2]; };

__device__ __forceinline__ void loadB(Frag (&b)[4], const unsigned short* p, int blo) {
#pragma unroll
    for (int c = 0; c < 4; ++c) {
        const unsigned short* q = p + 32 * c + blo;
        b[c].u[0] = *(const uint4*)(q);
        b[c].u[1] = *(const uint4*)(q + 8);
    }
}

__device__ __forceinline__ void tileCompute(const Frag (&a)[4], const Frag (&b)[4],
                                            const unsigned (&rh)[8], int ncol,
                                            float (&bestS)[8], unsigned (&bestI)[8]) {
    v8f acc = {0.f, 0.f, 0.f, 0.f, 0.f, 0.f, 0.f, 0.f};
#pragma unroll
    for (int c = 0; c < 4; ++c)
        acc = __builtin_amdgcn_wmma_f32_16x16x32_bf16(
                  false, a[c].v, false, b[c].v, (short)0, acc, false, false);

    // column hash: once per tile per lane
    unsigned nch = (unsigned)ncol * 0x9E3779B1u;
    nch ^= nch >> 16; nch *= 0x85EBCA77u;

#pragma unroll
    for (int r = 0; r < 8; ++r) {
        unsigned hsh = rh[r] ^ nch;
        hsh *= 0x2C1B3C6Du; hsh ^= hsh >> 17;
        float u = (float)(hsh >> 8) * 0x1.0p-24f + 0x1.0p-25f;      // (0,1), never denormal
        float e = -NLN2 * __builtin_amdgcn_logf(u);                 // -ln(u) > 0, raw v_log_f32
        float g = -NLN2 * __builtin_amdgcn_logf(e);                 // gumbel sample
        float s = acc[r] + g;
        bool  gt = s > bestS[r];
        bestS[r] = gt ? s : bestS[r];                               // v_cndmask, EXEC untouched
        bestI[r] = gt ? (unsigned)ncol : bestI[r];
    }
}

__global__ void __launch_bounds__(256) k_score_argmax(
        const unsigned short* __restrict__ hb,
        const unsigned short* __restrict__ itb,
        unsigned long long*   __restrict__ best) {
    const int lane   = threadIdx.x & 31;
    const int wave   = threadIdx.x >> 5;
    const int mgrp   = blockIdx.x & 15;     // 16 groups x 8 waves = 128 M-tiles
    const int nsplit = blockIdx.x >> 4;     // 125 N-splits
    const int m0     = (mgrp * 8 + wave) * 16;
    const int mhalf  = m0 + ((lane >= 16) ? 8 : 0);

    // --- A fragments: 16 rows of h (bf16), K=128, resident for the whole loop
    Frag a[4];
    const unsigned short* abase = hb + (unsigned)(m0 + (lane & 15)) * DFEAT;
    const int alo = (lane < 16) ? 0 : 16;
    const int ahi = (lane < 16) ? 8 : 24;
#pragma unroll
    for (int c = 0; c < 4; ++c) {
        a[c].u[0] = *(const uint4*)(abase + 32 * c + alo);
        a[c].u[1] = *(const uint4*)(abase + 32 * c + ahi);
    }

    // --- row hashes: full avalanche hoisted out of the N loop
    unsigned rh[8];
#pragma unroll
    for (int r = 0; r < 8; ++r) rh[r] = avalanche((unsigned)(mhalf + r) * 0x9E3779B1u);

    float    bestS[8];
    unsigned bestI[8];
#pragma unroll
    for (int r = 0; r < 8; ++r) { bestS[r] = -3.4e38f; bestI[r] = 0u; }

    const int t0  = nsplit * TILES_PER_SPLIT;
    const int blo = (lane < 16) ? 0 : 16;
    const unsigned short* rowptr = itb + (unsigned)(t0 * 16 + (lane & 15)) * DFEAT;
    int ncol = t0 * 16 + (lane & 15);

    // --- software-pipelined double buffer: 8 b128 loads always in flight ----
    Frag b0[4], b1[4];
    loadB(b0, rowptr, blo);                                  // tile 0
#pragma unroll 1
    for (int tp = 0; tp < TILES_PER_SPLIT / 2; ++tp) {       // 12 pairs
        loadB(b1, rowptr + 16 * DFEAT, blo);                 // tile 2tp+1
        __builtin_prefetch(rowptr + 48 * DFEAT, 0, 3);       // L2->near warm, 3 tiles ahead
        tileCompute(a, b0, rh, ncol, bestS, bestI);          // tile 2tp
        loadB(b0, rowptr + 32 * DFEAT, blo);                 // tile 2tp+2 (<= 24, in range)
        tileCompute(a, b1, rh, ncol + 16, bestS, bestI);     // tile 2tp+1
        rowptr += 32 * DFEAT;
        ncol   += 32;
    }
    tileCompute(a, b0, rh, ncol, bestS, bestI);              // tail tile 24

    // --- pack once, reduce across the 16 lanes of each half, one atomic/row --
#pragma unroll
    for (int r = 0; r < 8; ++r) {
        unsigned ub  = __float_as_uint(bestS[r]);
        unsigned key = (ub & 0x80000000u) ? ~ub : (ub | 0x80000000u);
        unsigned long long v = ((unsigned long long)key << 32) | bestI[r];
#pragma unroll
        for (int m = 1; m < 16; m <<= 1) {
            unsigned long long o = __shfl_xor(v, m, 32);     // stays within 16-lane halves
            v = (v > o) ? v : o;
        }
        if ((lane & 15) == 0)
            atomicMax(&best[mhalf + r], v);
    }
}

// ---------------------------------------------------------------------------
// Kernel 6: gather repl/orig vectors, cosine similarity, write repl_items
// ---------------------------------------------------------------------------
__global__ void k_sim(const unsigned long long* __restrict__ best,
                      const int* __restrict__ uii,
                      const float* __restrict__ items,
                      int* __restrict__ repl_out,
                      float* __restrict__ simbuf) {
    int row = blockIdx.x * blockDim.x + threadIdx.x;
    if (row >= BATCH) return;
    int idx  = (int)(best[row] & 0xFFFFFFFFull);
    repl_out[row] = idx;
    int orig = uii[row * 2 + 1];
    const float* f1 = items + (size_t)orig * DFEAT;
    const float* f2 = items + (size_t)idx  * DFEAT;
    float dot = 0.f, n1 = 0.f, n2 = 0.f;
#pragma unroll 8
    for (int k = 0; k < DFEAT; ++k) {
        float x = f1[k], y = f2[k];
        dot = fmaf(x, y, dot);
        n1  = fmaf(x, x, n1);
        n2  = fmaf(y, y, n2);
    }
    float d   = fmaxf(sqrtf(n1), COS_EPS_) * fmaxf(sqrtf(n2), COS_EPS_);
    float sim = dot / d;
    simbuf[row] = (sim + 1.f) * 0.5f;
}

// ---------------------------------------------------------------------------
// Kernel 7: deterministic fixed-tree reduction of loss and mean
// ---------------------------------------------------------------------------
__global__ void k_final(const float* __restrict__ simbuf, float* __restrict__ out2) {
    __shared__ float sl[256], sm[256];
    int t = threadIdx.x;
    float a = 0.f, b = 0.f;
    for (int r = t; r < BATCH; r += 256) {
        float s = simbuf[r];
        float d = s - 0.5f;                // labels are SIM_RATIO = 0.5
        a += d * d;
        b += s;
    }
    sl[t] = a; sm[t] = b;
    __syncthreads();
    for (int o = 128; o > 0; o >>= 1) {
        if (t < o) { sl[t] += sl[t + o]; sm[t] += sm[t + o]; }
        __syncthreads();
    }
    if (t == 0) {
        out2[0] = sl[0] * (1.f / BATCH);   // sim_loss
        out2[1] = sm[0] * (1.f / BATCH);   // sim.mean()
    }
}

// ---------------------------------------------------------------------------
extern "C" void kernel_launch(void* const* d_in, const int* in_sizes, int n_in,
                              void* d_out, int out_size, void* d_ws, size_t ws_size,
                              hipStream_t stream) {
    const int*   user_item_id = (const int*)  d_in[0];   // (2048, 2)
    const float* item_feature = (const float*)d_in[1];   // (2048, 256)
    const float* all_items    = (const float*)d_in[2];   // (50000, 128)
    const float* W            = (const float*)d_in[3];   // (256, 128)
    const float* bias         = (const float*)d_in[4];   // (128,)
    const float* gamma        = (const float*)d_in[5];   // (128,)
    const float* beta         = (const float*)d_in[6];   // (128,)

    char* ws = (char*)d_ws;
    float*              h_f32  = (float*)             (ws + WS_H_F32);
    unsigned short*     h_bf16 = (unsigned short*)    (ws + WS_H_BF16);
    unsigned short*     it_bf  = (unsigned short*)    (ws + WS_ITEMS_BF16);
    unsigned long long* best   = (unsigned long long*)(ws + WS_BEST);
    float*              simbuf = (float*)             (ws + WS_SIM);

    int*   repl_out = (int*)d_out;                 // first 2048: repl_items (int32)
    float* scal_out = ((float*)d_out) + BATCH;     // then sim_loss, sim_mean

    k_mlp<<<(BATCH * DFEAT) / 256, 256, 0, stream>>>(item_feature, W, bias, h_f32);
    k_bn<<<DFEAT, 256, 0, stream>>>(h_f32, gamma, beta, h_bf16);
    k_cvt_items<<<(NITEMS * DFEAT) / 256, 256, 0, stream>>>(all_items, it_bf);
    k_init<<<(BATCH + 255) / 256, 256, 0, stream>>>(best);
    k_score_argmax<<<16 * NSPLIT, 256, 0, stream>>>(h_bf16, it_bf, best);
    k_sim<<<(BATCH + 255) / 256, 256, 0, stream>>>(best, user_item_id, all_items,
                                                   repl_out, simbuf);
    k_final<<<1, 256, 0, stream>>>(simbuf, scal_out);
}
